// PerformerAttention_90640989815100
// MI455X (gfx1250) — matrix operand order repaired and checked by reading, not compiled, not run on hardware
//
#include <hip/hip_runtime.h>
#include <math.h>

// Problem constants (B=4, H=16 -> BH=64)
#define BH_N 64
#define T_N  4096
#define D_N  64
#define R_N  256
#define SPLIT 16                 // T-chunks per head in the kv pass
#define TCHUNK (T_N / SPLIT)     // 256
#define TT 32                    // t rows staged per iteration
#define LDK 72                   // padded LDS stride for D=64 tiles (bf16)
#define LDV 72
#define LDP 264                  // padded LDS stride for R=256 tiles (bf16)
#define EPS_F 1e-6f

typedef __attribute__((ext_vector_type(16))) __bf16 v16bf;
typedef __attribute__((ext_vector_type(2)))  __bf16 bf2;
typedef __attribute__((ext_vector_type(8)))  float  v8f;

#define WMMA_BF16(a, b, c) \
  __builtin_amdgcn_wmma_f32_16x16x32_bf16(false, (a), false, (b), (short)0, (c), false, false)

// ---- WMMA fragment gathers (layouts per CDNA5 ISA §7.12.2) ----

// A: 16x32 (MxK) bf16, row-major source src[m*ld + k]
__device__ __forceinline__ v16bf frag_a_lds(const __bf16* src, int ld, int lane) {
  v16bf a;
  const int hf = lane >> 4;
  const int m  = lane & 15;
  const __bf16* row = src + m * ld;
#pragma unroll
  for (int j = 0; j < 8; ++j) {
    const int kb = ((j & 4) << 2) + (hf << 3) + ((j & 3) << 1); // (j>=4?16:0)+half*8+2*(j%4)
    bf2 p = *(const bf2*)(row + kb);                            // 4B-aligned pair load
    a[2 * j]     = p[0];
    a[2 * j + 1] = p[1];
  }
  return a;
}

// A: 16x32 (MxK) logical, but source is stored K-major: A(m,k) = src[k*ld + m]
__device__ __forceinline__ v16bf frag_a_lds_T(const __bf16* src, int ld, int lane) {
  v16bf a;
  const int hf = lane >> 4;
  const int m  = lane & 15;
#pragma unroll
  for (int j = 0; j < 8; ++j) {
    const int kb = ((j & 4) << 2) + (hf << 3) + ((j & 3) << 1);
    a[2 * j]     = src[kb * ld + m];
    a[2 * j + 1] = src[(kb + 1) * ld + m];
  }
  return a;
}

// B: 32x16 (KxN) bf16, row-major source src[k*ld + n]
__device__ __forceinline__ v16bf frag_b_lds(const __bf16* src, int ld, int lane) {
  v16bf b;
  const int hf = lane >> 4;
  const int n  = lane & 15;
#pragma unroll
  for (int j = 0; j < 8; ++j) {
    const int k0 = ((j & 4) << 2) + (hf << 3) + ((j & 3) << 1); // (j>=4?16:0)+half*8+2*(j%4)
    b[2 * j]     = src[k0 * ld + n];
    b[2 * j + 1] = src[(k0 + 1) * ld + n];
  }
  return b;
}

// B: 32x16 (KxN) gathered from f32 global memory (converted to bf16)
__device__ __forceinline__ v16bf frag_b_gf32(const float* src, int ld, int lane) {
  v16bf b;
  const int hf = lane >> 4;
  const int n  = lane & 15;
#pragma unroll
  for (int j = 0; j < 8; ++j) {
    const int k0 = ((j & 4) << 2) + (hf << 3) + ((j & 3) << 1);
    b[2 * j]     = (__bf16)src[k0 * ld + n];
    b[2 * j + 1] = (__bf16)src[(k0 + 1) * ld + n];
  }
  return b;
}

// -------------------------- kernels --------------------------

__global__ void performer_zero_kernel(float* p, int n) {
  const int i = blockIdx.x * blockDim.x + threadIdx.x;
  if (i < n) p[i] = 0.f;
}

// Pass A: kv[bh,r,d] = sum_t k'(t,r) v(t,d);  ksum[bh,r] = sum_t k'(t,r)
__global__ __launch_bounds__(256) void performer_kv_kernel(
    const float* __restrict__ K, const float* __restrict__ V,
    const float* __restrict__ M, const float* __restrict__ W,
    float* __restrict__ KV, float* __restrict__ KS) {
  __shared__ __bf16 sK[TT * LDK];
  __shared__ __bf16 sV[TT * LDV];
  __shared__ __bf16 sP[TT * LDP];
  __shared__ float  sNorm[TT];
  __shared__ float  sMask[TT];
  __shared__ float  sSum[R_N];

  const int tid   = threadIdx.x;
  const int lane  = tid & 31;
  const int wv    = tid >> 5;              // 8 waves
  const int bh    = blockIdx.x / SPLIT;
  const int chunk = blockIdx.x % SPLIT;
  const int bIdx  = bh >> 4;               // H == 16
  const size_t base = (size_t)bh * T_N * D_N;
  const int tstart  = chunk * TCHUNK;

  for (int i = tid; i < R_N; i += 256) sSum[i] = 0.f;

  // Pre-gather W B-fragments from global (L2-resident, 64KB): wave wv owns r in [32wv, 32wv+32)
  v16bf wb[2][2];
#pragma unroll
  for (int kc = 0; kc < 2; ++kc)
#pragma unroll
    for (int rt = 0; rt < 2; ++rt)
      wb[kc][rt] = frag_b_gf32(W + (kc * 32) * R_N + (wv * 32 + rt * 16), R_N, lane);

  // kv accumulators: wave wv owns r tiles {2wv, 2wv+1} x all 4 d tiles
  v8f acc[2][4];
#pragma unroll
  for (int rt = 0; rt < 2; ++rt)
#pragma unroll
    for (int dt = 0; dt < 4; ++dt)
#pragma unroll
      for (int e = 0; e < 8; ++e) acc[rt][dt][e] = 0.f;

  const int row = tid >> 3;  // 32 rows
  const int seg = tid & 7;   // 8 f32 per thread

  for (int it = 0; it < TCHUNK / TT; ++it) {
    const int t0 = tstart + it * TT;

    // ---- stage k,v tiles + row norms + mask ----
    {
      const float* kp = K + base + (size_t)(t0 + row) * D_N + seg * 8;
      const float* vp = V + base + (size_t)(t0 + row) * D_N + seg * 8;
      const float mk  = M[bIdx * T_N + t0 + row];
      float ss = 0.f;
#pragma unroll
      for (int e = 0; e < 8; ++e) {
        const float kx = kp[e];
        ss += kx * kx;
        sK[row * LDK + seg * 8 + e] = (__bf16)kx;
        sV[row * LDV + seg * 8 + e] = (__bf16)(vp[e] * mk);
      }
      ss += __shfl_xor(ss, 1);
      ss += __shfl_xor(ss, 2);
      ss += __shfl_xor(ss, 4);
      if (seg == 0) { sNorm[row] = 0.5f * ss; sMask[row] = mk; }
    }
    __syncthreads();

    // ---- k' = exp(k@W - 0.5||k||^2)/sqrt(R) * mask  (WMMA proj + f32 exp) ----
#pragma unroll
    for (int tt = 0; tt < 2; ++tt) {
      v16bf a0 = frag_a_lds(&sK[(tt * 16) * LDK + 0],  LDK, lane);
      v16bf a1 = frag_a_lds(&sK[(tt * 16) * LDK + 32], LDK, lane);
#pragma unroll
      for (int rt = 0; rt < 2; ++rt) {
        v8f c = {};
        c = WMMA_BF16(a0, wb[0][rt], c);
        c = WMMA_BF16(a1, wb[1][rt], c);
        const int hf = lane >> 4, n = lane & 15;
#pragma unroll
        for (int e = 0; e < 8; ++e) {
          const int tr = tt * 16 + e + 8 * hf;
          const float val = __expf(c[e] - sNorm[tr]) * 0.0625f * sMask[tr];
          sP[tr * LDP + wv * 32 + rt * 16 + n] = (__bf16)val;
        }
      }
    }
    __syncthreads();

    // ---- k_sum column partials (thread tid owns column tid) ----
    {
      float s = 0.f;
#pragma unroll
      for (int t = 0; t < TT; ++t) s += (float)sP[t * LDP + tid];
      sSum[tid] += s;
    }

    // ---- kv += k'^T @ v  (A gathered transposed from LDS) ----
    {
      v16bf bv[4];
#pragma unroll
      for (int dt = 0; dt < 4; ++dt) bv[dt] = frag_b_lds(&sV[dt * 16], LDV, lane);
#pragma unroll
      for (int rt = 0; rt < 2; ++rt) {
        v16bf at = frag_a_lds_T(&sP[wv * 32 + rt * 16], LDP, lane);
#pragma unroll
        for (int dt = 0; dt < 4; ++dt)
          acc[rt][dt] = WMMA_BF16(at, bv[dt], acc[rt][dt]);
      }
    }
    __syncthreads();
  }

  // ---- commit partials with hardware f32 atomics ----
  float* kvp = KV + (size_t)bh * R_N * D_N;
  const int hf = lane >> 4, n = lane & 15;
#pragma unroll
  for (int rt = 0; rt < 2; ++rt)
#pragma unroll
    for (int dt = 0; dt < 4; ++dt)
#pragma unroll
      for (int e = 0; e < 8; ++e) {
        const int r = wv * 32 + rt * 16 + e + 8 * hf;
        const int d = dt * 16 + n;
        unsafeAtomicAdd(&kvp[r * D_N + d], acc[rt][dt][e]);
      }
  unsafeAtomicAdd(&KS[bh * R_N + tid], sSum[tid]);
}

// Pass B: out = (q' @ kv) / (q' @ ksum + eps)
__global__ __launch_bounds__(256) void performer_out_kernel(
    const float* __restrict__ Q, const float* __restrict__ W,
    const float* __restrict__ KV, const float* __restrict__ KS,
    float* __restrict__ O) {
  __shared__ __bf16 sQ[64 * LDK];
  __shared__ __bf16 sP[64 * LDP];
  __shared__ float  sKs[R_N];
  __shared__ float  sNorm[64];
  __shared__ float  sZ[64];

  const int tid = threadIdx.x, lane = tid & 31, wv = tid >> 5;
  const int bh = blockIdx.x >> 6;          // 64 q-tiles of 64 rows
  const int qt = blockIdx.x & 63;
  const size_t base = (size_t)bh * T_N * D_N + (size_t)qt * 64 * D_N;
  const float* kvp = KV + (size_t)bh * R_N * D_N;

  // ---- stage q tile (64x64) + norms ----
  {
    const int row = tid >> 2, seg = tid & 3;
    const float* qp = Q + base + row * D_N + seg * 16;
    float ss = 0.f;
#pragma unroll
    for (int e = 0; e < 16; ++e) {
      const float x = qp[e];
      ss += x * x;
      sQ[row * LDK + seg * 16 + e] = (__bf16)x;
    }
    ss += __shfl_xor(ss, 1);
    ss += __shfl_xor(ss, 2);
    if (seg == 0) sNorm[row] = 0.5f * ss;
  }
  for (int i = tid; i < R_N; i += 256) sKs[i] = KS[bh * R_N + i];

  v16bf wb[2][2];
#pragma unroll
  for (int kc = 0; kc < 2; ++kc)
#pragma unroll
    for (int rt = 0; rt < 2; ++rt)
      wb[kc][rt] = frag_b_gf32(W + (kc * 32) * R_N + (wv * 32 + rt * 16), R_N, lane);
  __syncthreads();

  // ---- q' tiles: wave wv covers r in [32wv, 32wv+32) for all 4 t sub-tiles ----
#pragma unroll
  for (int tt = 0; tt < 4; ++tt) {
    v16bf a0 = frag_a_lds(&sQ[(tt * 16) * LDK + 0],  LDK, lane);
    v16bf a1 = frag_a_lds(&sQ[(tt * 16) * LDK + 32], LDK, lane);
#pragma unroll
    for (int rt = 0; rt < 2; ++rt) {
      v8f c = {};
      c = WMMA_BF16(a0, wb[0][rt], c);
      c = WMMA_BF16(a1, wb[1][rt], c);
      const int hf = lane >> 4, n = lane & 15;
#pragma unroll
      for (int e = 0; e < 8; ++e) {
        const int tr = tt * 16 + e + 8 * hf;
        const float val = __expf(c[e] - sNorm[tr]) * 0.0625f;
        sP[tr * LDP + wv * 32 + rt * 16 + n] = (__bf16)val;
      }
    }
  }
  __syncthreads();

  // ---- z[row] = q'(row,:) . ksum  (4 threads/row, shfl reduce) ----
  {
    const int row = tid >> 2, part = tid & 3;
    float s = 0.f;
#pragma unroll 16
    for (int r = part * 64; r < part * 64 + 64; ++r)
      s += (float)sP[row * LDP + r] * sKs[r];
    s += __shfl_xor(s, 1);
    s += __shfl_xor(s, 2);
    if (part == 0) sZ[row] = s;
  }

  // ---- out tiles: wave wv -> d tile (wv&3), t tiles {(wv>>2)*2, +1}; K = 256 ----
  const int dt  = wv & 3;
  const int ttb = (wv >> 2) * 2;
  v8f o0 = {}, o1 = {};
#pragma unroll
  for (int kc = 0; kc < 8; ++kc) {
    v16bf bk = frag_b_gf32(kvp + (kc * 32) * D_N + dt * 16, D_N, lane);
    v16bf a0 = frag_a_lds(&sP[(ttb * 16) * LDP + kc * 32],       LDP, lane);
    v16bf a1 = frag_a_lds(&sP[((ttb + 1) * 16) * LDP + kc * 32], LDP, lane);
    o0 = WMMA_BF16(a0, bk, o0);
    o1 = WMMA_BF16(a1, bk, o1);
  }
  __syncthreads();  // sZ ready

  float* op = O + base;
  const int hf = lane >> 4, n = lane & 15;
#pragma unroll
  for (int e = 0; e < 8; ++e) {
    const int r0 = ttb * 16 + e + 8 * hf;
    const int r1 = (ttb + 1) * 16 + e + 8 * hf;
    op[r0 * D_N + dt * 16 + n] = o0[e] / (sZ[r0] + EPS_F);
    op[r1 * D_N + dt * 16 + n] = o1[e] / (sZ[r1] + EPS_F);
  }
}

// -------------------------- launch --------------------------

extern "C" void kernel_launch(void* const* d_in, const int* in_sizes, int n_in,
                              void* d_out, int out_size, void* d_ws, size_t ws_size,
                              hipStream_t stream) {
  const float* q  = (const float*)d_in[0];
  const float* k  = (const float*)d_in[1];
  const float* v  = (const float*)d_in[2];
  const float* m  = (const float*)d_in[3];
  const float* w  = (const float*)d_in[4];
  float* out = (float*)d_out;

  float* kv = (float*)d_ws;                          // BH*R*D f32 = 4 MB
  float* ks = kv + (size_t)BH_N * R_N * D_N;         // BH*R   f32 = 64 KB
  const int nz = BH_N * R_N * D_N + BH_N * R_N;

  performer_zero_kernel<<<(nz + 255) / 256, 256, 0, stream>>>(kv, nz);
  performer_kv_kernel<<<BH_N * SPLIT, 256, 0, stream>>>(k, v, m, w, kv, ks);
  performer_out_kernel<<<BH_N * (T_N / 64), 256, 0, stream>>>(q, w, kv, ks, out);
}
